// PowerfulCrossAttention_54348516164312
// MI455X (gfx1250) — matrix operand order repaired and checked by reading, not compiled, not run on hardware
//
#include <hip/hip_runtime.h>
#include <hip/hip_bf16.h>
#include <math.h>

typedef _Float16 half_t;
typedef _Float16 v16h __attribute__((ext_vector_type(16)));
typedef _Float16 v8h  __attribute__((ext_vector_type(8)));
typedef float    v8f  __attribute__((ext_vector_type(8)));
typedef float    f4   __attribute__((ext_vector_type(4)));

#define BATCH 32
#define CCH   256
#define HH    32
#define WW    32
#define NHD   8
#define HD    32
#define NTOK  16
#define NPIX  32768   /* B*H*W */
#define NWIN  2048    /* B * 8 * 8 */
#define HID   1024

// ---------------------------------------------------------------------------
// Wave-per-(16 x NB*16)-tile f16 WMMA GEMM core, NB-way N register blocking.
// K is a compile-time constant and the k-loop is fully unrolled: no phi/move
// overhead, and with __launch_bounds__(256,1) on the callers the scheduler
// has the VGPR budget to hoist each iteration's 10 b128 loads ahead of the
// previous iteration's WMMAs (partial s_wait_loadcnt, load/WMMA overlap).
// A row-major [M,K] f16, W row-major [N,K] f16 (B^T rows == W rows).
// A fragment: lane m = lane&15, halves = K[hi*16..+16).
// B fragment: lane n = lane&15, halves = K[hi*16..+16).
// ---------------------------------------------------------------------------
template <int NB, int K>
__device__ __forceinline__ void gemm_tileN(const half_t* __restrict__ A,
                                           const half_t* __restrict__ W,
                                           int row_base, int col_base,
                                           int lane, v8f acc[NB])
{
    const int l15 = lane & 15;
    const int hi  = lane >> 4;
    const half_t* arow = A + (size_t)(row_base + l15) * K + hi * 16;
    const half_t* wrow[NB];
    #pragma unroll
    for (int t = 0; t < NB; ++t)
        wrow[t] = W + (size_t)(col_base + t * 16 + l15) * K + hi * 16;
    #pragma unroll
    for (int t = 0; t < NB; ++t) acc[t] = (v8f){};
    #pragma unroll
    for (int k0 = 0; k0 < K; k0 += 32) {
        if (k0 + 32 < K) __builtin_prefetch(arow + k0 + 32, 0, 3);
        v8h  alo = *(const v8h*)(arow + k0);
        v8h  ahi = *(const v8h*)(arow + k0 + 8);
        v16h a;
        #pragma unroll
        for (int i = 0; i < 8; ++i) { a[i] = alo[i]; a[i + 8] = ahi[i]; }
        #pragma unroll
        for (int t = 0; t < NB; ++t) {
            v16h b = *(const v16h*)(wrow[t] + k0);
            acc[t] = __builtin_amdgcn_wmma_f32_16x16x32_f16(false, a, false, b,
                                                            (short)0, acc[t],
                                                            false, false);
        }
    }
}

// ---------------------------------------------------------------------------
// K0: LayerNorm1 + NCHW->NHWC transpose; emits f32 x_hwc (residual) and f16
// window-partitioned LN output hwin[NWIN*16, 256].
// One block per (b,h) row: 32 pixels x 256 channels staged through LDS.
// ---------------------------------------------------------------------------
__global__ void k_ln1(const float* __restrict__ x, const float* __restrict__ g,
                      const float* __restrict__ bt, float* __restrict__ x_hwc,
                      half_t* __restrict__ hwin)
{
    __shared__ float tile[256][33];
    const int bh = blockIdx.x;            // b*32 + h
    const int b = bh >> 5, h = bh & 31;
    const int c = threadIdx.x;
    const float* src = x + (((size_t)(b * CCH + c)) * HH + h) * WW;
    #pragma unroll
    for (int w4 = 0; w4 < 32; w4 += 4) {
        f4 v = *(const f4*)(src + w4);
        tile[c][w4 + 0] = v[0]; tile[c][w4 + 1] = v[1];
        tile[c][w4 + 2] = v[2]; tile[c][w4 + 3] = v[3];
    }
    __syncthreads();
    const int p  = threadIdx.x >> 3;      // pixel (w) 0..31
    const int gq = threadIdx.x & 7;       // channel group: 32 channels each
    float s = 0.f, ss = 0.f;
    #pragma unroll
    for (int i = 0; i < 32; ++i) {
        float v = tile[gq * 32 + i][p];
        s += v; ss += v * v;
    }
    #pragma unroll
    for (int msk = 1; msk <= 4; msk <<= 1) {
        s  += __shfl_xor(s,  msk, 32);
        ss += __shfl_xor(ss, msk, 32);
    }
    const float mean = s * (1.f / 256.f);
    const float inv  = rsqrtf(ss * (1.f / 256.f) - mean * mean + 1e-5f);
    const int w = p;
    const size_t pix = (size_t)bh * 32 + w;
    const int rwin = ((b * 8 + (h >> 2)) * 8 + (w >> 2)) * 16 + (h & 3) * 4 + (w & 3);
    float*  xo = x_hwc + pix * CCH + gq * 32;
    half_t* ho = hwin + (size_t)rwin * CCH + gq * 32;
    #pragma unroll
    for (int i = 0; i < 32; ++i) {
        float v = tile[gq * 32 + i][p];
        xo[i] = v;
        ho[i] = (half_t)((v - mean) * inv * g[gq * 32 + i] + bt[gq * 32 + i]);
    }
}

// ---------------------------------------------------------------------------
// K1: QKV GEMM  [32768,256] x [768,256]^T + bias -> f16 [32768,768]
// ---------------------------------------------------------------------------
__global__ void __launch_bounds__(256, 1)
k_qkv(const half_t* __restrict__ hwin, const half_t* __restrict__ wq,
      const float* __restrict__ bias, half_t* __restrict__ qkv)
{
    const int wid  = (blockIdx.x * blockDim.x + threadIdx.x) >> 5;
    const int lane = threadIdx.x & 31;
    const int mt = wid / 12, ng = wid % 12;
    v8f acc[4];
    gemm_tileN<4, CCH>(hwin, wq, mt * 16, ng * 64, lane, acc);
    const int n = lane & 15, hi = lane >> 4;
    #pragma unroll
    for (int t = 0; t < 4; ++t) {
        const int col = ng * 64 + t * 16 + n;
        const float bv = bias[col];
        #pragma unroll
        for (int r = 0; r < 8; ++r) {
            const int m = r + hi * 8;
            qkv[(size_t)(mt * 16 + m) * 768 + col] = (half_t)(acc[t][r] + bv);
        }
    }
}

// ---------------------------------------------------------------------------
// K2: window attention. One wave per (window, head).
// S = q k^T via one WMMA (K=hd=32); softmax across lanes; P*V via two WMMAs
// with K padded 16->32 (zero halves / zero B rows). P and V^T staged in LDS.
// ---------------------------------------------------------------------------
__global__ void __launch_bounds__(256, 1)
k_attn(const half_t* __restrict__ qkv, const float* __restrict__ rpb,
       half_t* __restrict__ attn_out)
{
    __shared__ __align__(32) half_t ldsP[8][16 * 16];
    __shared__ __align__(32) half_t ldsV[8][32 * 16];
    const int wv   = threadIdx.x >> 5;
    const int wid  = blockIdx.x * 8 + wv;
    const int lane = threadIdx.x & 31;
    const int win = wid >> 3, head = wid & 7;
    const int l15 = lane & 15, hi = lane >> 4;
    const half_t* base = qkv + (size_t)win * 16 * 768;

    // Q fragment (A layout)
    v16h qa;
    {
        const half_t* qrow = base + l15 * 768 + head * 32 + hi * 16;
        v8h qlo = *(const v8h*)(qrow);
        v8h qhi = *(const v8h*)(qrow + 8);
        #pragma unroll
        for (int i = 0; i < 8; ++i) { qa[i] = qlo[i]; qa[i + 8] = qhi[i]; }
    }
    // K fragment (B layout): B^T rows == K-matrix rows (row-major)
    v16h kb = *(const v16h*)(base + l15 * 768 + 256 + head * 32 + hi * 16);
    v8f c0 = {};
    v8f S = __builtin_amdgcn_wmma_f32_16x16x32_f16(false, qa, false, kb,
                                                   (short)0, c0, false, false);

    // logits + rel-pos bias + softmax (rows spread across 16-lane groups)
    const float scale = 0.17677669529663687f;  // 1/sqrt(32)
    const int n = l15, ni = n >> 2, nj = n & 3;
    float p[8];
    #pragma unroll
    for (int r = 0; r < 8; ++r) {
        const int m = r + hi * 8;
        const int idx = ((m >> 2) - ni + 3) * 7 + ((m & 3) - nj + 3);
        p[r] = S[r] * scale + rpb[idx * 8 + head];
    }
    #pragma unroll
    for (int r = 0; r < 8; ++r) {
        float mx = p[r];
        #pragma unroll
        for (int msk = 1; msk <= 8; msk <<= 1) mx = fmaxf(mx, __shfl_xor(mx, msk, 32));
        const float e = expf(p[r] - mx);
        float sm = e;
        #pragma unroll
        for (int msk = 1; msk <= 8; msk <<= 1) sm += __shfl_xor(sm, msk, 32);
        p[r] = e / sm;
    }
    // Stage P (C layout -> row-major) and V^T in LDS
    #pragma unroll
    for (int r = 0; r < 8; ++r)
        ldsP[wv][(r + hi * 8) * 16 + n] = (half_t)p[r];
    {
        v16h vv = *(const v16h*)(base + l15 * 768 + 512 + head * 32 + hi * 16);
        #pragma unroll
        for (int i = 0; i < 16; ++i)
            ldsV[wv][(hi * 16 + i) * 16 + l15] = vv[i];
    }
    __syncthreads();

    // A = P padded to 16x32 (upper K halves zero)
    v16h pa;
    {
        v16h pfull = *(const v16h*)(&ldsP[wv][l15 * 16]);
        #pragma unroll
        for (int i = 0; i < 16; ++i) pa[i] = (hi == 0) ? pfull[i] : (half_t)0;
    }
    #pragma unroll
    for (int t = 0; t < 2; ++t) {
        v16h vb;
        if (hi == 0) {
            vb = *(const v16h*)(&ldsV[wv][(t * 16 + l15) * 16]);
        } else {
            #pragma unroll
            for (int i = 0; i < 16; ++i) vb[i] = (half_t)0;  // K rows 16..31
        }
        v8f cz = {};
        v8f o = __builtin_amdgcn_wmma_f32_16x16x32_f16(false, pa, false, vb,
                                                       (short)0, cz, false, false);
        #pragma unroll
        for (int r = 0; r < 8; ++r) {
            const int m = r + hi * 8;
            attn_out[(size_t)(win * 16 + m) * CCH + head * 32 + t * 16 + n] = (half_t)o[r];
        }
    }
}

// ---------------------------------------------------------------------------
// K3: proj GEMM + window-reverse + residual1 -> f32 y[pixel, C]
// ---------------------------------------------------------------------------
__global__ void __launch_bounds__(256, 1)
k_proj(const half_t* __restrict__ ao, const half_t* __restrict__ wp,
       const float* __restrict__ bias, const float* __restrict__ x_hwc,
       float* __restrict__ y)
{
    const int wid  = (blockIdx.x * blockDim.x + threadIdx.x) >> 5;
    const int lane = threadIdx.x & 31;
    const int mt = wid / 4, ng = wid % 4;
    v8f acc[4];
    gemm_tileN<4, CCH>(ao, wp, mt * 16, ng * 64, lane, acc);
    const int n = lane & 15, hi = lane >> 4;
    #pragma unroll
    for (int t = 0; t < 4; ++t) {
        const int col = ng * 64 + t * 16 + n;
        const float bv = bias[col];
        #pragma unroll
        for (int r = 0; r < 8; ++r) {
            const int rw = mt * 16 + r + hi * 8;
            const int win = rw >> 4, tok = rw & 15;
            const int b = win >> 6, wh = (win >> 3) & 7, ww = win & 7;
            const size_t pix = ((size_t)b * 32 + wh * 4 + (tok >> 2)) * 32 + ww * 4 + (tok & 3);
            const size_t off = pix * CCH + col;
            y[off] = x_hwc[off] + acc[t][r] + bv;
        }
    }
}

// ---------------------------------------------------------------------------
// K4: LayerNorm2 — one wave per pixel-row of y, output f16 h2.
// ---------------------------------------------------------------------------
__global__ void k_ln2(const float* __restrict__ y, const float* __restrict__ g,
                      const float* __restrict__ bt, half_t* __restrict__ h2)
{
    const int wid  = (blockIdx.x * blockDim.x + threadIdx.x) >> 5;
    const int lane = threadIdx.x & 31;
    const float* row = y + (size_t)wid * CCH + lane * 8;
    f4 v0 = *(const f4*)(row);
    f4 v1 = *(const f4*)(row + 4);
    float s = 0.f, ss = 0.f;
    #pragma unroll
    for (int i = 0; i < 4; ++i) { s += v0[i] + v1[i]; ss += v0[i]*v0[i] + v1[i]*v1[i]; }
    #pragma unroll
    for (int msk = 1; msk <= 16; msk <<= 1) {
        s  += __shfl_xor(s,  msk, 32);
        ss += __shfl_xor(ss, msk, 32);
    }
    const float mean = s * (1.f / 256.f);
    const float inv  = rsqrtf(ss * (1.f / 256.f) - mean * mean + 1e-5f);
    half_t* out = h2 + (size_t)wid * CCH + lane * 8;
    #pragma unroll
    for (int i = 0; i < 4; ++i) {
        out[i]     = (half_t)((v0[i] - mean) * inv * g[lane*8 + i]     + bt[lane*8 + i]);
        out[i + 4] = (half_t)((v1[i] - mean) * inv * g[lane*8 + i + 4] + bt[lane*8 + i + 4]);
    }
}

// ---------------------------------------------------------------------------
// K5: MLP fc1 + exact GELU -> f16 G[32768, 1024]
// ---------------------------------------------------------------------------
__global__ void __launch_bounds__(256, 1)
k_mlp1(const half_t* __restrict__ h2, const half_t* __restrict__ w1,
       const float* __restrict__ b1, half_t* __restrict__ G)
{
    const int wid  = (blockIdx.x * blockDim.x + threadIdx.x) >> 5;
    const int lane = threadIdx.x & 31;
    const int mt = wid / 16, ng = wid % 16;
    v8f acc[4];
    gemm_tileN<4, CCH>(h2, w1, mt * 16, ng * 64, lane, acc);
    const int n = lane & 15, hi = lane >> 4;
    #pragma unroll
    for (int t = 0; t < 4; ++t) {
        const int col = ng * 64 + t * 16 + n;
        const float bv = b1[col];
        #pragma unroll
        for (int r = 0; r < 8; ++r) {
            float v = acc[t][r] + bv;
            v = 0.5f * v * (1.f + erff(v * 0.70710678118654752f));
            G[(size_t)(mt * 16 + r + hi * 8) * HID + col] = (half_t)v;
        }
    }
}

// ---------------------------------------------------------------------------
// K6: MLP fc2 + residual2 + NHWC->NCHW store to d_out (f32)
// ---------------------------------------------------------------------------
__global__ void __launch_bounds__(256, 1)
k_mlp2(const half_t* __restrict__ G, const half_t* __restrict__ w2,
       const float* __restrict__ b2, const float* __restrict__ y,
       float* __restrict__ out)
{
    const int wid  = (blockIdx.x * blockDim.x + threadIdx.x) >> 5;
    const int lane = threadIdx.x & 31;
    const int mt = wid / 4, ng = wid % 4;
    v8f acc[4];
    gemm_tileN<4, HID>(G, w2, mt * 16, ng * 64, lane, acc);
    const int n = lane & 15, hi = lane >> 4;
    #pragma unroll
    for (int t = 0; t < 4; ++t) {
        const int col = ng * 64 + t * 16 + n;
        const float bv = b2[col];
        #pragma unroll
        for (int r = 0; r < 8; ++r) {
            const size_t pix = (size_t)(mt * 16 + r + hi * 8);
            const float v = y[pix * CCH + col] + acc[t][r] + bv;
            const int b = (int)(pix >> 10), h = (int)((pix >> 5) & 31), w = (int)(pix & 31);
            out[(((size_t)b * CCH + col) * HH + h) * WW + w] = v;
        }
    }
}

// f32 -> f16 weight conversion (grid-stride)
__global__ void k_cvt(const float* __restrict__ s, half_t* __restrict__ d, int n)
{
    for (int i = blockIdx.x * blockDim.x + threadIdx.x; i < n; i += gridDim.x * blockDim.x)
        d[i] = (half_t)s[i];
}

extern "C" void kernel_launch(void* const* d_in, const int* in_sizes, int n_in,
                              void* d_out, int out_size, void* d_ws, size_t ws_size,
                              hipStream_t stream)
{
    const float* x       = (const float*)d_in[0];
    const float* n1g     = (const float*)d_in[1];
    const float* n1b     = (const float*)d_in[2];
    const float* qkv_w   = (const float*)d_in[3];
    const float* qkv_b   = (const float*)d_in[4];
    const float* rpb     = (const float*)d_in[5];
    const float* proj_w  = (const float*)d_in[6];
    const float* proj_b  = (const float*)d_in[7];
    const float* n2g     = (const float*)d_in[8];
    const float* n2b     = (const float*)d_in[9];
    const float* mlp_w1  = (const float*)d_in[10];
    const float* mlp_b1  = (const float*)d_in[11];
    const float* mlp_w2  = (const float*)d_in[12];
    const float* mlp_b2  = (const float*)d_in[13];
    float* out = (float*)d_out;

    char* p = (char*)d_ws;
    float*  x_hwc = (float*)p;   p += (size_t)NPIX * CCH * 4;   // 32 MB
    float*  yres  = (float*)p;   p += (size_t)NPIX * CCH * 4;   // 32 MB
    half_t* hwin  = (half_t*)p;  p += (size_t)NPIX * CCH * 2;   // 16 MB (reused as h2)
    half_t* qkv   = (half_t*)p;  p += (size_t)NPIX * HID * 2;   // 64 MB (qkv then MLP hidden)
    half_t* aout  = (half_t*)p;  p += (size_t)NPIX * CCH * 2;   // 16 MB
    half_t* wq    = (half_t*)p;  p += (size_t)768 * 256 * 2;
    half_t* wp    = (half_t*)p;  p += (size_t)256 * 256 * 2;
    half_t* w1    = (half_t*)p;  p += (size_t)HID * 256 * 2;
    half_t* w2    = (half_t*)p;  p += (size_t)256 * HID * 2;
    half_t* h2    = hwin;        // reuse: hwin dead after K1
    half_t* G     = qkv;         // reuse: qkv dead after K2

    // weight conversions
    k_cvt<<<384, 256, 0, stream>>>(qkv_w,  wq, 768 * 256);
    k_cvt<<<128, 256, 0, stream>>>(proj_w, wp, 256 * 256);
    k_cvt<<<512, 256, 0, stream>>>(mlp_w1, w1, HID * 256);
    k_cvt<<<512, 256, 0, stream>>>(mlp_w2, w2, 256 * HID);

    k_ln1 <<<BATCH * HH,        256, 0, stream>>>(x, n1g, n1b, x_hwc, hwin);
    k_qkv <<<NWIN * 12 / 8,     256, 0, stream>>>(hwin, wq, qkv_b, qkv);   // 24576 waves
    k_attn<<<NWIN * NHD / 8,    256, 0, stream>>>(qkv, rpb, aout);         // 16384 waves
    k_proj<<<NWIN * 4 / 8,      256, 0, stream>>>(aout, wp, proj_b, x_hwc, yres);
    k_ln2 <<<NPIX / 8,          256, 0, stream>>>(yres, n2g, n2b, h2);
    k_mlp1<<<NWIN * 16 / 8,     256, 0, stream>>>(h2, w1, mlp_b1, G);      // 32768 waves
    k_mlp2<<<NWIN * 4 / 8,      256, 0, stream>>>(G, w2, mlp_b2, yres, out);
}
